// Transformer_48696339202592
// MI455X (gfx1250) — matrix operand order repaired and checked by reading, not compiled, not run on hardware
//
#include <hip/hip_runtime.h>
#include <hip/hip_bf16.h>

// ---------------------------------------------------------------------------
// MI455X / gfx1250, wave32. Matrix ops: v_wmma_f32_16x16x32_bf16.
// ---------------------------------------------------------------------------

typedef __attribute__((ext_vector_type(16))) __bf16 v16bf;
typedef __attribute__((ext_vector_type(8)))  float  v8f;

#define NN 50000            // nodes
#define NE 800000           // edges
#define HIDDIM 64
#define BN_SCALE 0.9999950000374997f   // 1/sqrt(1+1e-5)

__device__ __forceinline__ float selu_f(float x) {
    const float lam = 1.0507009873554805f, alp = 1.6732632423543772f;
    return x > 0.f ? lam * x : lam * alp * (__expf(x) - 1.f);
}

// ---------------------------------------------------------------------------
// Collapse  de -> W_de -> Wpd[l] -> head-colsum  (and m path) into tiny mats.
// ew layout per layer (stride 128 floats): Wcd[16][4] | Wcm[8][4] @64 | bs[4] @96
// ---------------------------------------------------------------------------
__global__ void precompute_edge_kernel(
    const float* __restrict__ W_de, const float* __restrict__ b_de,
    const float* __restrict__ W_m,  const float* __restrict__ b_m,
    const float* __restrict__ Wpd,  const float* __restrict__ bpd,
    const float* __restrict__ Wpm,  const float* __restrict__ bpm,
    float* __restrict__ ew)
{
    __shared__ float cs_d[64][4];
    __shared__ float cs_m[64][4];
    const int t = threadIdx.x;   // 64 threads
    for (int l = 0; l < 2; ++l) {
        const float* Wpd_l = Wpd + l * 64 * 64;
        const float* Wpm_l = Wpm + l * 64 * 64;
        const float* bpd_l = bpd + l * 64;
        const float* bpm_l = bpm + l * 64;
        float* ewl = ew + l * 128;
        if (t < 64) {
            for (int hh = 0; hh < 4; ++hh) {
                float a = 0.f, b = 0.f;
                for (int j = 0; j < 16; ++j) {
                    a += Wpd_l[t * 64 + hh * 16 + j];
                    b += Wpm_l[t * 64 + hh * 16 + j];
                }
                cs_d[t][hh] = a; cs_m[t][hh] = b;
            }
        }
        __syncthreads();
        if (t < 16) {
            for (int hh = 0; hh < 4; ++hh) {
                float a = 0.f;
                for (int k = 0; k < 64; ++k) a += W_de[t * 64 + k] * cs_d[k][hh];
                ewl[t * 4 + hh] = a;
            }
        }
        if (t < 8) {
            for (int hh = 0; hh < 4; ++hh) {
                float a = 0.f;
                for (int k = 0; k < 64; ++k) a += W_m[t * 64 + k] * cs_m[k][hh];
                ewl[64 + t * 4 + hh] = a;
            }
        }
        if (t < 4) {
            float a = 0.f;
            for (int k = 0; k < 64; ++k) a += b_de[k] * cs_d[k][t] + b_m[k] * cs_m[k][t];
            for (int j = 0; j < 16; ++j) a += bpd_l[t * 16 + j] + bpm_l[t * 16 + j];
            ewl[96 + t] = a;
        }
        __syncthreads();
    }
}

// ---------------------------------------------------------------------------
// WMMA GEMM with compile-time shapes:
//   C[rows,NC] = act( A[rows,K] @ W[K,NC] + bias (+ res) ),  rows % 16 == 0.
// Fragment layouts per CDNA5 ISA 7.12.2 (wave32):
//   A 16x32 bf16: lane m=lane&15, hi=lane>>4;
//     elems 0..7  -> K = kstep*32 + 8*hi  + 0..7   (contiguous -> 2x float4)
//     elems 8..15 -> K = kstep*32 + 16 + 8*hi + 0..7 (contiguous -> 2x float4)
//   B 32x16 bf16: lane n=lane&15, hi=lane>>4; elem e -> K = 16*(e>>3) + 8*hi + (e&7).
//     W is pre-swizzled into LDS in fragment order so one lane's 16 bf16 are a
//     single 32-byte aligned contiguous LDS read.
//   C 16x16 f32 : elem r -> row = 8*hi + r, col = lane&15
// ACT: 0=none 1=relu 2=selu
// ---------------------------------------------------------------------------
template<int K, int NC, int ACT>
__global__ __launch_bounds__(256) void gemm_wmma_kernel(
    const float* __restrict__ A, const float* __restrict__ W,
    const float* __restrict__ bias, const float* __restrict__ res,
    float* __restrict__ C, int rows)
{
    constexpr int KSTEPS = (K + 31) / 32;
    constexpr int NT = NC / 16;
    constexpr int SW_ELEMS = KSTEPS * NT * 512;     // <= 32KB bf16
    __shared__ __bf16 sW[SW_ELEMS];
    __shared__ float  sB[NC];
    const int tid = threadIdx.x;

    // Cooperative fill of W into pre-swizzled B-fragment order.
    for (int idx = tid; idx < SW_ELEMS; idx += 256) {
        const int e     = idx & 15;
        const int lane_ = (idx >> 4) & 31;
        const int rest  = idx >> 9;
        const int t     = rest % NT;
        const int ks    = rest / NT;
        const int k     = ks * 32 + ((e >> 3) << 4) + ((lane_ >> 4) << 3) + (e & 7);
        const int n     = t * 16 + (lane_ & 15);
        sW[idx] = (k < K) ? (__bf16)W[k * NC + n] : (__bf16)0.f;
    }
    for (int i = tid; i < NC; i += 256) sB[i] = bias ? bias[i] : 0.f;
    __syncthreads();

    const int lane = tid & 31, wave = tid >> 5;
    const int hi = lane >> 4, ln = lane & 15;
    const int tileCount = rows >> 4;

    for (int tile = blockIdx.x * 8 + wave; tile < tileCount; tile += gridDim.x * 8) {
        v8f acc[NT];
        #pragma unroll
        for (int t = 0; t < NT; ++t)
            #pragma unroll
            for (int r = 0; r < 8; ++r) acc[t][r] = 0.f;

        const float* Arow = A + (size_t)(tile * 16 + ln) * K;   // 64B+ aligned
        #pragma unroll
        for (int ks = 0; ks < KSTEPS; ++ks) {
            v16bf af;
            {
                const int b0 = ks * 32 + 8 * hi;                // always < K
                const float4 p0 = *(const float4*)(Arow + b0);
                const float4 p1 = *(const float4*)(Arow + b0 + 4);
                af[0] = (__bf16)p0.x; af[1] = (__bf16)p0.y;
                af[2] = (__bf16)p0.z; af[3] = (__bf16)p0.w;
                af[4] = (__bf16)p1.x; af[5] = (__bf16)p1.y;
                af[6] = (__bf16)p1.z; af[7] = (__bf16)p1.w;
                if constexpr (K % 32 == 0) {
                    const int b1 = ks * 32 + 16 + 8 * hi;
                    const float4 p2 = *(const float4*)(Arow + b1);
                    const float4 p3 = *(const float4*)(Arow + b1 + 4);
                    af[8]  = (__bf16)p2.x; af[9]  = (__bf16)p2.y;
                    af[10] = (__bf16)p2.z; af[11] = (__bf16)p2.w;
                    af[12] = (__bf16)p3.x; af[13] = (__bf16)p3.y;
                    af[14] = (__bf16)p3.z; af[15] = (__bf16)p3.w;
                } else {                                        // K == 16 pad
                    #pragma unroll
                    for (int e = 8; e < 16; ++e) af[e] = (__bf16)0.f;
                }
            }
            #pragma unroll
            for (int t = 0; t < NT; ++t) {
                const v16bf bf = *(const v16bf*)(sW + ((ks * NT + t) * 32 + lane) * 16);
                acc[t] = __builtin_amdgcn_wmma_f32_16x16x32_bf16(
                    false, af, false, bf, (short)0, acc[t], false, false);
            }
        }
        #pragma unroll
        for (int t = 0; t < NT; ++t) {
            const int n = t * 16 + ln;
            #pragma unroll
            for (int r = 0; r < 8; ++r) {
                const int row = tile * 16 + 8 * hi + r;
                float v = acc[t][r] + sB[n];
                if (res) v += res[(size_t)row * NC + n];
                if (ACT == 1) v = fmaxf(v, 0.f);
                if (ACT == 2) v = selu_f(v);
                C[(size_t)row * NC + n] = v;
            }
        }
    }
}

// ---------------------------------------------------------------------------
// Edge attention: one thread per edge; collapsed score weights; f32 atomics
// into wV[N,64] / z[N,4] (dst sorted -> atomics stay hot in L2).
// s = exp(clip(4*dot(K[src]_h, Q[dst]_h) + s0[h], -5, 5))
// ---------------------------------------------------------------------------
__global__ __launch_bounds__(256) void edge_attn_kernel(
    const int* __restrict__ src, const int* __restrict__ dst,
    const float* __restrict__ de, const float* __restrict__ m,
    const float* __restrict__ Q, const float* __restrict__ Kh,
    const float* __restrict__ V, const float* __restrict__ ew,
    float* __restrict__ wV, float* __restrict__ z, int nE)
{
    const int e = blockIdx.x * blockDim.x + threadIdx.x;
    if (e >= nE) return;
    const int sn = src[e], dn = dst[e];
    const float4* k4 = (const float4*)(Kh + (size_t)sn * HIDDIM);   // 256B rows
    const float4* q4 = (const float4*)(Q  + (size_t)dn * HIDDIM);
    const float4* v4 = (const float4*)(V  + (size_t)sn * HIDDIM);
    __builtin_prefetch(k4, 0, 0);   // -> global_prefetch_b8
    __builtin_prefetch(q4, 0, 0);
    __builtin_prefetch(v4, 0, 0);

    float s0[4];
    #pragma unroll
    for (int h = 0; h < 4; ++h) s0[h] = ew[96 + h];
    const float4* de4 = (const float4*)(de + (size_t)e * 16);
    #pragma unroll
    for (int q = 0; q < 4; ++q) {
        const float4 d4 = de4[q];
        const float dv[4] = { d4.x, d4.y, d4.z, d4.w };
        #pragma unroll
        for (int j = 0; j < 4; ++j)
            #pragma unroll
            for (int h = 0; h < 4; ++h) s0[h] += dv[j] * ew[(q * 4 + j) * 4 + h];
    }
    const float4* m4 = (const float4*)(m + (size_t)e * 8);
    #pragma unroll
    for (int q = 0; q < 2; ++q) {
        const float4 d4 = m4[q];
        const float dv[4] = { d4.x, d4.y, d4.z, d4.w };
        #pragma unroll
        for (int j = 0; j < 4; ++j)
            #pragma unroll
            for (int h = 0; h < 4; ++h) s0[h] += dv[j] * ew[64 + (q * 4 + j) * 4 + h];
    }
    #pragma unroll
    for (int h = 0; h < 4; ++h) {
        float dot = 0.f;
        #pragma unroll
        for (int q = 0; q < 4; ++q) {
            const float4 ka = k4[h * 4 + q];
            const float4 qa = q4[h * 4 + q];
            dot += ka.x * qa.x + ka.y * qa.y + ka.z * qa.z + ka.w * qa.w;
        }
        float sc = 4.f * dot + s0[h];                 // 16 * dot * (1/sqrt(16))
        sc = __expf(fminf(fmaxf(sc, -5.f), 5.f));
        atomicAdd(&z[(size_t)dn * 4 + h], sc);
        #pragma unroll
        for (int q = 0; q < 4; ++q) {
            const float4 va = v4[h * 4 + q];
            float* base = wV + (size_t)dn * HIDDIM + h * 16 + q * 4;
            atomicAdd(base + 0, sc * va.x);
            atomicAdd(base + 1, sc * va.y);
            atomicAdd(base + 2, sc * va.z);
            atomicAdd(base + 3, sc * va.w);
        }
    }
}

__global__ void attn_combine_kernel(const float* __restrict__ wV,
                                    const float* __restrict__ z,
                                    float* __restrict__ out, int n)
{
    const int i = blockIdx.x * blockDim.x + threadIdx.x;
    if (i >= n) return;
    const int node = i >> 6, h = (i & 63) >> 4;
    out[i] = wV[i] / (z[node * 4 + h] + 1e-10f);
}

__global__ void fill_zero_kernel(float* __restrict__ p, int n)
{
    const int i = blockIdx.x * blockDim.x + threadIdx.x;
    if (i < n) p[i] = 0.f;
}

// ---------------------------------------------------------------------------
// LayerNorm over 64 cols, one wave32 per row, shfl_xor reduction; folds the
// eval-mode BatchNorm scale.  out = ((x-mu)*rsqrt(var+eps)*g + b) * BN_SCALE
// ---------------------------------------------------------------------------
__global__ __launch_bounds__(256) void ln_kernel(
    const float* __restrict__ in, const float* __restrict__ g,
    const float* __restrict__ b, float* __restrict__ out, int rows)
{
    const int lane = threadIdx.x & 31;
    const int row = blockIdx.x * 8 + (threadIdx.x >> 5);
    if (row >= rows) return;
    const float* x = in + (size_t)row * HIDDIM;
    const int c = 2 * lane;
    float2 v = *(const float2*)(x + c);
    float s = v.x + v.y;
    float sq = v.x * v.x + v.y * v.y;
    #pragma unroll
    for (int o = 16; o >= 1; o >>= 1) {
        s  += __shfl_xor(s,  o, 32);
        sq += __shfl_xor(sq, o, 32);
    }
    const float mu  = s * (1.f / 64.f);
    const float var = sq * (1.f / 64.f) - mu * mu;
    const float inv = rsqrtf(var + 1e-5f);
    float2 o2;
    o2.x = ((v.x - mu) * inv * g[c]     + b[c])     * BN_SCALE;
    o2.y = ((v.y - mu) * inv * g[c + 1] + b[c + 1]) * BN_SCALE;
    *(float2*)(out + (size_t)row * HIDDIM + c) = o2;
}

// ---------------------------------------------------------------------------
extern "C" void kernel_launch(void* const* d_in, const int* in_sizes, int n_in,
                              void* d_out, int out_size, void* d_ws, size_t ws_size,
                              hipStream_t stream)
{
    (void)in_sizes; (void)n_in; (void)out_size; (void)ws_size;

    const int*   src  = (const int*)d_in[0];
    const int*   dst  = (const int*)d_in[1];
    const float* x    = (const float*)d_in[2];
    const float* pe   = (const float*)d_in[3];
    const float* de   = (const float*)d_in[4];
    const float* m    = (const float*)d_in[5];
    const float* I    = (const float*)d_in[6];
    const float* W_h  = (const float*)d_in[7];
    const float* W_pe = (const float*)d_in[8];
    const float* b_pe = (const float*)d_in[9];
    const float* W_de = (const float*)d_in[10];
    const float* b_de = (const float*)d_in[11];
    const float* W_m  = (const float*)d_in[12];
    const float* b_m  = (const float*)d_in[13];
    const float* Wq   = (const float*)d_in[14];
    const float* bq   = (const float*)d_in[15];
    const float* Wk   = (const float*)d_in[16];
    const float* bk   = (const float*)d_in[17];
    const float* Wv   = (const float*)d_in[18];
    const float* bv   = (const float*)d_in[19];
    const float* Wpd  = (const float*)d_in[20];
    const float* bpd  = (const float*)d_in[21];
    const float* Wpm  = (const float*)d_in[22];
    const float* bpm  = (const float*)d_in[23];
    const float* WO   = (const float*)d_in[24];
    const float* bO   = (const float*)d_in[25];
    const float* Wpi  = (const float*)d_in[26];
    const float* bpi  = (const float*)d_in[27];
    const float* g1   = (const float*)d_in[28];
    const float* b1   = (const float*)d_in[29];
    const float* g2   = (const float*)d_in[30];
    const float* b2   = (const float*)d_in[31];
    const float* Wf1  = (const float*)d_in[32];
    const float* bf1  = (const float*)d_in[33];
    const float* Wf2  = (const float*)d_in[34];
    const float* bf2  = (const float*)d_in[35];
    const float* Wr1  = (const float*)d_in[36];
    const float* br1  = (const float*)d_in[37];
    const float* Wr2  = (const float*)d_in[38];
    const float* br2  = (const float*)d_in[39];

    float* ws = (float*)d_ws;
    const size_t NH = (size_t)NN * HIDDIM;          // 3.2M floats
    float* h    = ws; ws += NH;
    float* Qb   = ws; ws += NH;
    float* Kb   = ws; ws += NH;
    float* Vb   = ws; ws += NH;
    float* wV   = ws; ws += NH;
    float* z    = ws; ws += (size_t)NN * 4;         // contiguous after wV
    float* hmid = ws; ws += NH;
    float* hout = ws; ws += NH;
    float* ew   = ws; ws += 256;
    float* f    = Qb;   // [N,128] aliases Q+K (dead during FFN phase)
    float* r    = Vb;   // [N,16]  aliases V   (dead during readout)

    float* out_h = (float*)d_out;
    float* out_x = out_h + NH;

    const dim3 blk(256);
    const int gemmGrid = (NN / 16 + 7) / 8;               // 391 blocks (8 waves each)
    const int edgeGrid = (NE + 255) / 256;
    const int lnGrid   = (NN + 7) / 8;
    const int zeroN    = NN * (HIDDIM + 4);               // wV + z together
    const int combGrid = ((int)NH + 255) / 256;

    precompute_edge_kernel<<<1, 64, 0, stream>>>(W_de, b_de, W_m, b_m, Wpd, bpd, Wpm, bpm, ew);

    // h = x @ W_h  ;  h += pe @ W_pe + b_pe
    gemm_wmma_kernel<128, 64, 0><<<gemmGrid, blk, 0, stream>>>(x,  W_h,  nullptr, nullptr, h, NN);
    gemm_wmma_kernel<16,  64, 0><<<gemmGrid, blk, 0, stream>>>(pe, W_pe, b_pe,    h,       h, NN);

    for (int l = 0; l < 2; ++l) {
        const float* Wq_l  = Wq  + l * 64 * 64;  const float* bq_l  = bq  + l * 64;
        const float* Wk_l  = Wk  + l * 64 * 64;  const float* bk_l  = bk  + l * 64;
        const float* Wv_l  = Wv  + l * 64 * 64;  const float* bv_l  = bv  + l * 64;
        const float* WO_l  = WO  + l * 64 * 64;  const float* bO_l  = bO  + l * 64;
        const float* Wpi_l = Wpi + l * 16 * 64;  const float* bpi_l = bpi + l * 64;
        const float* Wf1_l = Wf1 + l * 64 * 128; const float* bf1_l = bf1 + l * 128;
        const float* Wf2_l = Wf2 + l * 128 * 64; const float* bf2_l = bf2 + l * 64;

        gemm_wmma_kernel<64, 64, 0><<<gemmGrid, blk, 0, stream>>>(h, Wq_l, bq_l, nullptr, Qb, NN);
        gemm_wmma_kernel<64, 64, 0><<<gemmGrid, blk, 0, stream>>>(h, Wk_l, bk_l, nullptr, Kb, NN);
        gemm_wmma_kernel<64, 64, 0><<<gemmGrid, blk, 0, stream>>>(h, Wv_l, bv_l, nullptr, Vb, NN);

        fill_zero_kernel<<<(zeroN + 255) / 256, blk, 0, stream>>>(wV, zeroN);
        edge_attn_kernel<<<edgeGrid, blk, 0, stream>>>(src, dst, de, m, Qb, Kb, Vb,
                                                       ew + l * 128, wV, z, NE);
        attn_combine_kernel<<<combGrid, blk, 0, stream>>>(wV, z, hmid, (int)NH);

        // hmid += I @ Wpi + bpi ;  hout = hmid @ WO + bO + h (residual)
        gemm_wmma_kernel<16, 64, 0><<<gemmGrid, blk, 0, stream>>>(I,    Wpi_l, bpi_l, hmid, hmid, NN);
        gemm_wmma_kernel<64, 64, 0><<<gemmGrid, blk, 0, stream>>>(hmid, WO_l,  bO_l,  h,    hout, NN);
        ln_kernel<<<lnGrid, blk, 0, stream>>>(hout, g1 + l * 64, b1 + l * 64, h, NN);

        // FFN with residual folded, then LN2 (last layer writes straight to d_out)
        gemm_wmma_kernel<64, 128, 1><<<gemmGrid, blk, 0, stream>>>(h, Wf1_l, bf1_l, nullptr, f, NN);
        gemm_wmma_kernel<128, 64, 0><<<gemmGrid, blk, 0, stream>>>(f, Wf2_l, bf2_l, h,       hout, NN);
        ln_kernel<<<lnGrid, blk, 0, stream>>>(hout, g2 + l * 64, b2 + l * 64,
                                              (l == 1) ? out_h : h, NN);
    }

    // x_hat = selu(h @ Wr1 + br1) @ Wr2 + br2
    gemm_wmma_kernel<64, 16, 2><<<gemmGrid, blk, 0, stream>>>(out_h, Wr1, br1, nullptr, r,     NN);
    gemm_wmma_kernel<16, 128, 0><<<gemmGrid, blk, 0, stream>>>(r,     Wr2, br2, nullptr, out_x, NN);
}
